// ImageMLP_47029891891239
// MI455X (gfx1250) — 8-bit matrix operand order repaired and checked by reading, compiled, not run on hardware
//
#include <hip/hip_runtime.h>
#include <hip/hip_bf16.h>

typedef int v8i __attribute__((ext_vector_type(8)));

#define K_TOTAL   76800
#define B_ROWS    1024
#define N_OUT     8
#define MTILES    (B_ROWS / 16)
#define NKSPLIT   25
#define WG_KCHUNK (K_TOTAL / NKSPLIT)
#define WAVE_KCHK (WG_KCHUNK / 8)
#define KI        (WAVE_KCHK / 64)

#define OFF_ROWMAX 0
#define OFF_WPART  8192
#define OFF_WMEAN  12288
#define OFF_WQ8    16384
#define OFF_PART   630784

__device__ __forceinline__ int clampi(int v, int lo, int hi) {
    return v < lo ? lo : (v > hi ? hi : v);
}

__device__ __forceinline__ int quant_pack4(float4 f, float s) {
    int q0 = __float2int_rn(fminf(fmaxf(f.x * s, -128.0f), 127.0f));
    int q1 = __float2int_rn(fminf(fmaxf(f.y * s, -128.0f), 127.0f));
    int q2 = __float2int_rn(fminf(fmaxf(f.z * s, -128.0f), 127.0f));
    int q3 = __float2int_rn(fminf(fmaxf(f.w * s, -128.0f), 127.0f));
    return (int)(((unsigned)q0 & 255u) | (((unsigned)q1 & 255u) << 8) |
                 (((unsigned)q2 & 255u) << 16) | (((unsigned)q3 & 255u) << 24));
}

__global__ __launch_bounds__(256) void k_rowmax(const float* __restrict__ x,
                                                float* __restrict__ rowmax) {
    __shared__ float red[256];
    const int b = blockIdx.x;
    const float4* row = (const float4*)(x + (size_t)b * K_TOTAL);
    float m = 0.0f;
    for (int i = threadIdx.x; i < K_TOTAL / 4; i += 256) {
        float4 v = row[i];
        m = fmaxf(m, fmaxf(fmaxf(fabsf(v.x), fabsf(v.y)),
                           fmaxf(fabsf(v.z), fabsf(v.w))));
    }
    red[threadIdx.x] = m;
    __syncthreads();
    for (int s = 128; s > 0; s >>= 1) {
        if (threadIdx.x < s)
            red[threadIdx.x] = fmaxf(red[threadIdx.x], red[threadIdx.x + s]);
        __syncthreads();
    }
    if (threadIdx.x == 0) rowmax[b] = red[0];
}

__global__ __launch_bounds__(256) void k_wabs_partial(const float* __restrict__ W,
                                                      float* __restrict__ wpart) {
    __shared__ float red[256];
    const size_t base = (size_t)blockIdx.x * 2048 + (size_t)threadIdx.x * 8;
    const float4* p = (const float4*)(W + base);
    float4 a = p[0], b = p[1];
    float s = fabsf(a.x) + fabsf(a.y) + fabsf(a.z) + fabsf(a.w)
            + fabsf(b.x) + fabsf(b.y) + fabsf(b.z) + fabsf(b.w);
    red[threadIdx.x] = s;
    __syncthreads();
    for (int st = 128; st > 0; st >>= 1) {
        if (threadIdx.x < st) red[threadIdx.x] += red[threadIdx.x + st];
        __syncthreads();
    }
    if (threadIdx.x == 0) wpart[blockIdx.x] = red[0];
}

__global__ __launch_bounds__(256) void k_wmean(const float* __restrict__ wpart,
                                               float* __restrict__ wmean) {
    __shared__ float red[256];
    float s = 0.0f;
    for (int i = threadIdx.x; i < 300; i += 256) s += wpart[i];
    red[threadIdx.x] = s;
    __syncthreads();
    for (int st = 128; st > 0; st >>= 1) {
        if (threadIdx.x < st) red[threadIdx.x] += red[threadIdx.x + st];
        __syncthreads();
    }
    if (threadIdx.x == 0)
        wmean[0] = fmaxf(red[0] / 614400.0f, 1e-5f);
}

__global__ __launch_bounds__(256) void k_wquant(const float* __restrict__ W,
                                                const float* __restrict__ wmean,
                                                unsigned* __restrict__ wq8) {
    const float sw = 1.0f / wmean[0];
    const size_t i = (size_t)blockIdx.x * 256 + threadIdx.x;
    const float4 v = ((const float4*)W)[i];
    int q0 = clampi(__float2int_rn(v.x * sw), -1, 1);
    int q1 = clampi(__float2int_rn(v.y * sw), -1, 1);
    int q2 = clampi(__float2int_rn(v.z * sw), -1, 1);
    int q3 = clampi(__float2int_rn(v.w * sw), -1, 1);
    unsigned p = ((unsigned)q0 & 255u) | (((unsigned)q1 & 255u) << 8)
               | (((unsigned)q2 & 255u) << 16) | (((unsigned)q3 & 255u) << 24);
    wq8[i] = p;
}

__global__ __launch_bounds__(256) void k_gemm(const float* __restrict__ x,
                                              const unsigned char* __restrict__ wq8,
                                              const float* __restrict__ rowmax,
                                              int* __restrict__ part) {
    __shared__ int lds[8 * 256];
    const int tid  = threadIdx.x;
    const int lane = tid & 31;
    const int wave = tid >> 5;
    const int tile = blockIdx.x;
    const int ks   = blockIdx.y;
    const int hi   = (lane >> 4) & 1;

    const int mrow = tile * 16 + (lane & 15);
    const float rm = fmaxf(rowmax[mrow], 1e-5f);
    const float sA = 127.0f / rm;
    const float* arow = x + (size_t)mrow * K_TOTAL;

    const int n = lane & 15;
    const unsigned char* bcol = wq8 + (size_t)(n < N_OUT ? n : 0) * K_TOTAL;
    const unsigned bmask = (n < N_OUT) ? 0xFFFFFFFFu : 0u;

    int aofs[8], bofs[8];
#pragma unroll
    for (int j = 0; j < 8; ++j) {
        aofs[j] = (hi ? 16 : 0) + ((j & 3) << 2) + ((j & 4) << 3);
        bofs[j] = (hi ? 16 : 0) + ((j & 3) << 2) + ((j & 4) << 3);
    }

    const int kbase = ks * WG_KCHUNK + wave * WAVE_KCHK;

    v8i bmat[KI];
#pragma unroll
    for (int kk = 0; kk < KI; ++kk) {
        const int k0 = kbase + kk * 64;
#pragma unroll
        for (int j = 0; j < 8; ++j) {
            unsigned bw = *(const unsigned*)(bcol + k0 + bofs[j]);
            bmat[kk][j] = (int)(bw & bmask);
        }
    }

    v8i c0 = {0, 0, 0, 0, 0, 0, 0, 0};
    v8i c1 = {0, 0, 0, 0, 0, 0, 0, 0};

#pragma unroll
    for (int kk = 0; kk < KI; ++kk) {
        const int k0 = kbase + kk * 64;
        __builtin_prefetch((const void*)(arow + k0 + 128), 0, 0);
        v8i a;
#pragma unroll
        for (int j = 0; j < 8; ++j) {
            float4 f = *(const float4*)(arow + k0 + aofs[j]);
            a[j] = quant_pack4(f, sA);
        }
        if (kk & 1)
            c1 = __builtin_amdgcn_wmma_i32_16x16x64_iu8(true, a, true, bmat[kk], c1,
                                                        false, false);
        else
            c0 = __builtin_amdgcn_wmma_i32_16x16x64_iu8(true, a, true, bmat[kk], c0,
                                                        false, false);
    }

#pragma unroll
    for (int r = 0; r < 8; ++r) lds[wave * 256 + lane * 8 + r] = c0[r] + c1[r];
    __syncthreads();
    int sum = 0;
#pragma unroll
    for (int w = 0; w < 8; ++w) sum += lds[w * 256 + tid];
    part[((size_t)(ks * MTILES + tile)) * 256 + tid] = sum;
}

__global__ __launch_bounds__(256) void k_finalize(const int* __restrict__ part,
                                                  const float* __restrict__ rowmax,
                                                  const float* __restrict__ wmean,
                                                  float* __restrict__ out) {
    __shared__ float yv[16][8];
    const int tid  = threadIdx.x;
    const int tile = blockIdx.x;
    int sum = 0;
#pragma unroll
    for (int ks = 0; ks < NKSPLIT; ++ks)
        sum += part[((size_t)(ks * MTILES + tile)) * 256 + tid];

    const int lane = tid >> 3;
    const int r    = tid & 7;
    const int m    = r + ((lane >= 16) ? 8 : 0);
    const int n    = lane & 15;
    const int row  = tile * 16 + m;

    const float wm = wmean[0];
    const float rm = fmaxf(rowmax[row], 1e-5f);
    const float y  = (float)sum * (rm * (1.0f / 127.0f)) * wm;

    if (n < N_OUT) yv[m][n] = y;
    __syncthreads();
    if (n < N_OUT) {
        float mx = yv[m][0];
#pragma unroll
        for (int j = 1; j < N_OUT; ++j) mx = fmaxf(mx, yv[m][j]);
        float denom = 0.0f;
#pragma unroll
        for (int j = 0; j < N_OUT; ++j) denom += __expf(yv[m][j] - mx);
        out[(size_t)row * N_OUT + n] = __expf(y - mx) / denom;
    }
}

extern "C" void kernel_launch(void* const* d_in, const int* in_sizes, int n_in,
                              void* d_out, int out_size, void* d_ws, size_t ws_size,
                              hipStream_t stream) {
    const float* x = (const float*)d_in[0];
    const float* W = (const float*)d_in[1];
    float* out = (float*)d_out;

    char* ws = (char*)d_ws;
    float* rowmax        = (float*)(ws + OFF_ROWMAX);
    float* wpart         = (float*)(ws + OFF_WPART);
    float* wmean         = (float*)(ws + OFF_WMEAN);
    unsigned char* wq8   = (unsigned char*)(ws + OFF_WQ8);
    int*   part          = (int*)(ws + OFF_PART);

    k_rowmax<<<B_ROWS, 256, 0, stream>>>(x, rowmax);
    k_wabs_partial<<<300, 256, 0, stream>>>(W, wpart);
    k_wmean<<<1, 256, 0, stream>>>(wpart, wmean);
    k_wquant<<<600, 256, 0, stream>>>(W, wmean, (unsigned*)wq8);
    dim3 g3(MTILES, NKSPLIT, 1);
    k_gemm<<<g3, 256, 0, stream>>>(x, wq8, rowmax, part);
    k_finalize<<<MTILES, 256, 0, stream>>>(part, rowmax, wmean, out);
}
